// CrossAtt_block_75015898792084
// MI455X (gfx1250) — compile-verified, matrix operand order checked by reading, not run on hardware
//
#include <hip/hip_runtime.h>
#include <hip/hip_bf16.h>

typedef float  v8f   __attribute__((ext_vector_type(8)));
typedef __bf16 v16bf __attribute__((ext_vector_type(16)));

#define BM 128
#define BN 128
#define BK 32
#define LPAD 8   // LDS row padding (bf16 elems); rows = 80B -> 16B aligned

union FragU {
    uint4 u[2];
    v16bf v;
};

// ---------------------------------------------------------------------------
// Generic bf16 WMMA GEMM: C[M,Nc] = A[M,K] * W[K,Nc] (+bias) (+epilogue)
// Weights are passed PRE-TRANSPOSED as BwT[Nc][K] so both A and B tiles are
// K-contiguous in global memory: staging = b128 loads -> b128 LDS stores.
// Software-pipelined double-buffered LDS staging hides global latency
// behind the WMMA chain.
// MODE 0: outb = bf16(acc + bias)
// MODE 1: outf = resid + acc + bias           (fp32 residual write)
// MODE 2: outb = bf16(gelu(acc + bias))       (tanh-approx gelu)
// ---------------------------------------------------------------------------
template <int MODE>
__global__ __launch_bounds__(256) void gemm_bf16_wmma(
    const __bf16* __restrict__ A, const __bf16* __restrict__ BwT,
    const float* __restrict__ bias, const float* __restrict__ resid,
    float* __restrict__ outf, __bf16* __restrict__ outb,
    int M, int K, int Nc)
{
    __shared__ __align__(16) __bf16 As[2][BM][BK + LPAD];
    __shared__ __align__(16) __bf16 Bt[2][BN][BK + LPAD];

    const int tid  = threadIdx.x;
    const int lane = tid & 31;
    const int wid  = tid >> 5;      // 8 waves (wave32)
    const int wm   = wid >> 1;      // 0..3 -> m offset 32*wm
    const int wn   = wid & 1;       // 0..1 -> n offset 64*wn

    const int block_m = blockIdx.y * BM;
    const int block_n = blockIdx.x * BN;

    // ---- staging geometry: each thread owns two 8-element chunks/tile ----
    const int r0 = (tid * 8) / BK,         c0 = (tid * 8) % BK;
    const int r1 = ((256 + tid) * 8) / BK, c1 = ((256 + tid) * 8) % BK;
    int gA0 = block_m + r0; if (gA0 >= M) gA0 = M - 1;
    int gA1 = block_m + r1; if (gA1 >= M) gA1 = M - 1;
    const __bf16* pA0 = A   + (size_t)gA0 * K + c0;
    const __bf16* pA1 = A   + (size_t)gA1 * K + c1;
    const __bf16* pB0 = BwT + (size_t)(block_n + r0) * K + c0;
    const __bf16* pB1 = BwT + (size_t)(block_n + r1) * K + c1;

    uint4 ra0, ra1, rb0, rb1;
    auto ld = [&](int k0) {
        ra0 = *(const uint4*)(pA0 + k0);
        ra1 = *(const uint4*)(pA1 + k0);
        rb0 = *(const uint4*)(pB0 + k0);
        rb1 = *(const uint4*)(pB1 + k0);
    };
    auto st = [&](int buf) {
        *(uint4*)(&As[buf][r0][c0]) = ra0;
        *(uint4*)(&As[buf][r1][c1]) = ra1;
        *(uint4*)(&Bt[buf][r0][c0]) = rb0;
        *(uint4*)(&Bt[buf][r1][c1]) = rb1;
    };

    v8f acc[2][4] = {};
    const int kTiles = K / BK;
    const int kbase  = (lane < 16) ? 0 : 8;
    const int lan16  = lane & 15;

    ld(0);
    st(0);
    __syncthreads();

    for (int kt = 0; kt < kTiles; ++kt) {
        const int buf = kt & 1;

        // issue next tile's global loads early (latency hidden by WMMAs)
        if (kt + 1 < kTiles) ld((kt + 1) * BK);

        // ---- fragments: two aligned b128 LDS loads each ----
        v16bf afrag[2];
        #pragma unroll
        for (int mi = 0; mi < 2; ++mi) {
            int m = wm * 32 + mi * 16 + lan16;
            FragU f;
            f.u[0] = *(const uint4*)(&As[buf][m][kbase]);
            f.u[1] = *(const uint4*)(&As[buf][m][kbase + 16]);
            afrag[mi] = f.v;
        }
        v16bf bfrag[4];
        #pragma unroll
        for (int ni = 0; ni < 4; ++ni) {
            int n = wn * 64 + ni * 16 + lan16;
            FragU f;
            f.u[0] = *(const uint4*)(&Bt[buf][n][kbase]);
            f.u[1] = *(const uint4*)(&Bt[buf][n][kbase + 16]);
            bfrag[ni] = f.v;
        }

        #pragma unroll
        for (int mi = 0; mi < 2; ++mi)
            #pragma unroll
            for (int ni = 0; ni < 4; ++ni)
                acc[mi][ni] = __builtin_amdgcn_wmma_f32_16x16x32_bf16(
                    false, afrag[mi], false, bfrag[ni],
                    (short)0, acc[mi][ni], false, false);

        // store next tile into the other LDS buffer
        if (kt + 1 < kTiles) st((kt + 1) & 1);
        __syncthreads();
    }

    // ---- epilogue: lane holds column n, VGPR r -> row (r + 8*(lane>=16)) ----
    #pragma unroll
    for (int mi = 0; mi < 2; ++mi) {
        #pragma unroll
        for (int ni = 0; ni < 4; ++ni) {
            int n     = block_n + wn * 64 + ni * 16 + lan16;
            int mbase = block_m + wm * 32 + mi * 16 + ((lane < 16) ? 0 : 8);
            float bcol = bias[n];
            v8f a = acc[mi][ni];
            #pragma unroll
            for (int r = 0; r < 8; ++r) {
                int m = mbase + r;
                if (m < M) {
                    size_t idx = (size_t)m * Nc + n;
                    float val = a[r] + bcol;
                    if (MODE == 0) {
                        outb[idx] = (__bf16)val;
                    } else if (MODE == 1) {
                        outf[idx] = resid[idx] + val;
                    } else {
                        float t = 0.7978845608028654f *
                                  (val + 0.044715f * val * val * val);
                        outb[idx] = (__bf16)(0.5f * val * (1.0f + tanhf(t)));
                    }
                }
            }
        }
    }
}

// ---------------------------------------------------------------------------
// LayerNorm over C=256 (fp32 in, bf16 out). One wave32 per row.
// ---------------------------------------------------------------------------
__global__ __launch_bounds__(256) void ln_rows_kernel(
    const float* __restrict__ X, const float* __restrict__ g,
    const float* __restrict__ bta, __bf16* __restrict__ out, int R)
{
    int row  = blockIdx.x * 8 + (threadIdx.x >> 5);
    int lane = threadIdx.x & 31;
    if (row >= R) return;
    const float* x = X + (size_t)row * 256;
    float v[8], s = 0.f, s2 = 0.f;
    #pragma unroll
    for (int i = 0; i < 8; ++i) {
        float t = x[lane + 32 * i];
        v[i] = t; s += t; s2 += t * t;
    }
    #pragma unroll
    for (int m = 16; m >= 1; m >>= 1) {
        s  += __shfl_xor(s,  m, 32);
        s2 += __shfl_xor(s2, m, 32);
    }
    float mean = s * (1.f / 256.f);
    float var  = s2 * (1.f / 256.f) - mean * mean;
    float rstd = rsqrtf(var + 1e-5f);
    __bf16* o = out + (size_t)row * 256;
    #pragma unroll
    for (int i = 0; i < 8; ++i) {
        int c = lane + 32 * i;
        o[c] = (__bf16)((v[i] - mean) * rstd * g[c] + bta[c]);
    }
}

// ---------------------------------------------------------------------------
// Per-head LayerNorm over hd=32, in place on bf16 [B*N, 8*32]. Wave per row.
// ---------------------------------------------------------------------------
__global__ __launch_bounds__(256) void head_ln_kernel(
    __bf16* __restrict__ X, const float* __restrict__ g,
    const float* __restrict__ bta, int R)   // R = B*N*8
{
    int row  = blockIdx.x * 8 + (threadIdx.x >> 5);
    int lane = threadIdx.x & 31;
    if (row >= R) return;
    int token = row >> 3, h = row & 7;
    size_t base = (size_t)token * 256 + h * 32;
    float t = (float)X[base + lane];
    float s = t, s2 = t * t;
    #pragma unroll
    for (int m = 16; m >= 1; m >>= 1) {
        s  += __shfl_xor(s,  m, 32);
        s2 += __shfl_xor(s2, m, 32);
    }
    float mean = s * (1.f / 32.f);
    float var  = s2 * (1.f / 32.f) - mean * mean;
    float rstd = rsqrtf(var + 1e-5f);
    X[base + lane] = (__bf16)((t - mean) * rstd * g[h * 32 + lane] + bta[h * 32 + lane]);
}

// kv[b,h,d,e] += sum_t K[b,t,h,d] * V[b,t,h,e]  (token-chunked, atomic accum)
__global__ __launch_bounds__(256) void kv_kernel(
    const __bf16* __restrict__ Kb, const __bf16* __restrict__ Vb,
    float* __restrict__ kvf, int Ntok, int S)
{
    int s  = blockIdx.x % S;
    int bh = blockIdx.x / S;          // b*8 + h
    int b  = bh >> 3, h = bh & 7;
    int e  = threadIdx.x & 31;
    int dq = threadIdx.x >> 5;        // d block of 4
    float acc[4] = {0.f, 0.f, 0.f, 0.f};
    int t0 = s * 128;
    int t1 = t0 + 128; if (t1 > Ntok) t1 = Ntok;
    for (int t = t0; t < t1; ++t) {
        size_t base = ((size_t)b * Ntok + t) * 256 + h * 32;
        float vv = (float)Vb[base + e];
        #pragma unroll
        for (int j = 0; j < 4; ++j)
            acc[j] += (float)Kb[base + dq * 4 + j] * vv;
    }
    float* dst = kvf + (size_t)bh * 32 * 32;
    #pragma unroll
    for (int j = 0; j < 4; ++j)
        atomicAdd(&dst[(dq * 4 + j) * 32 + e], acc[j]);
}

// attn[b,t,h*32+e] = (sum_d q[b,t,h,d] * kv[b,h,d,e]) / N   (head-concat fused)
__global__ __launch_bounds__(256) void attn_kernel(
    const __bf16* __restrict__ Qb, const float* __restrict__ kvf,
    __bf16* __restrict__ attnb, int Ntok)
{
    size_t idx   = (size_t)blockIdx.x * 256 + threadIdx.x;
    size_t total = (size_t)8 * Ntok * 256;
    if (idx >= total) return;
    int col    = (int)(idx & 255);
    size_t row = idx >> 8;                 // b*Ntok + t
    int b = (int)(row / Ntok);
    int h = col >> 5, e = col & 31;
    const __bf16* q  = Qb + row * 256 + h * 32;
    const float*  kv = kvf + ((size_t)(b * 8 + h) * 32) * 32 + e;
    float acc = 0.f;
    #pragma unroll
    for (int d = 0; d < 32; ++d)
        acc += (float)q[d] * kv[d * 32];
    attnb[idx] = (__bf16)(acc / (float)Ntok);
}

// out[n*K + k] = bf16(in[k*Nc + n])  : fp32 -> bf16 + transpose (one-shot)
__global__ void convT_bf16_kernel(const float* __restrict__ in,
                                  __bf16* __restrict__ out, int K, int Nc)
{
    int i = blockIdx.x * 256 + threadIdx.x;
    if (i >= K * Nc) return;
    int n = i / K, k = i % K;
    out[i] = (__bf16)in[(size_t)k * Nc + n];
}

__global__ void zero_f32_kernel(float* __restrict__ p, int n)
{
    int i = blockIdx.x * 256 + threadIdx.x;
    if (i < n) p[i] = 0.f;
}

// ---------------------------------------------------------------------------
extern "C" void kernel_launch(void* const* d_in, const int* in_sizes, int n_in,
                              void* d_out, int out_size, void* d_ws, size_t ws_size,
                              hipStream_t stream)
{
    (void)in_sizes; (void)n_in; (void)out_size; (void)ws_size;
    constexpr int B = 8, Ntok = 85 * 85, C = 256, Ch = 1024;
    const int M = B * Ntok;                       // 57800

    const float* fx    = (const float*)d_in[0];
    const float* ln1_g = (const float*)d_in[1];
    const float* ln1_b = (const float*)d_in[2];
    const float* Wq    = (const float*)d_in[3];
    const float* bq    = (const float*)d_in[4];
    const float* Wk    = (const float*)d_in[5];
    const float* bk    = (const float*)d_in[6];
    const float* Wv    = (const float*)d_in[7];
    const float* bv    = (const float*)d_in[8];
    const float* Wo    = (const float*)d_in[9];
    const float* bo    = (const float*)d_in[10];
    const float* kln_g = (const float*)d_in[11];
    const float* kln_b = (const float*)d_in[12];
    const float* vln_g = (const float*)d_in[13];
    const float* vln_b = (const float*)d_in[14];
    const float* ln2_g = (const float*)d_in[15];
    const float* ln2_b = (const float*)d_in[16];
    const float* W1    = (const float*)d_in[17];
    const float* b1    = (const float*)d_in[18];
    const float* W2    = (const float*)d_in[19];
    const float* b2    = (const float*)d_in[20];

    // ---- workspace layout (bytes) ----
    const size_t BNC_bf = (size_t)M * C * 2;      // 29,593,600
    const size_t BNC_f  = (size_t)M * C * 4;
    char* ws = (char*)d_ws;
    __bf16* xb    = (__bf16*)(ws);                       // also x2b later
    __bf16* qb    = (__bf16*)(ws + 1 * BNC_bf);
    __bf16* kb    = (__bf16*)(ws + 2 * BNC_bf);
    __bf16* vb    = (__bf16*)(ws + 3 * BNC_bf);
    __bf16* attnb = (__bf16*)(ws + 4 * BNC_bf);
    float*  y1    = (float*) (ws + 5 * BNC_bf);
    char*   p     = ws + 5 * BNC_bf + BNC_f;
    float*  kvf   = (float*)p;            p += (size_t)B * 8 * 32 * 32 * 4;
    __bf16* wqb   = (__bf16*)p;           p += (size_t)C * C * 2;   // Wq^T
    __bf16* wkb   = (__bf16*)p;           p += (size_t)C * C * 2;   // Wk^T
    __bf16* wvb   = (__bf16*)p;           p += (size_t)C * C * 2;   // Wv^T
    __bf16* wob   = (__bf16*)p;           p += (size_t)C * C * 2;   // Wo^T
    __bf16* w1b   = (__bf16*)p;           p += (size_t)C * Ch * 2;  // W1^T [Ch][C]
    __bf16* w2b   = (__bf16*)p;                                     // W2^T [C][Ch]
    __bf16* hb    = qb;                   // [M,Ch] bf16 aliases q/k/v/attn region
    __bf16* x2b   = xb;

    // ---- 1. weights fp32 -> bf16, pre-transposed to [Nc][K] ----
    int ncc = C * C, nch = C * Ch;
    convT_bf16_kernel<<<(ncc + 255) / 256, 256, 0, stream>>>(Wq, wqb, C, C);
    convT_bf16_kernel<<<(ncc + 255) / 256, 256, 0, stream>>>(Wk, wkb, C, C);
    convT_bf16_kernel<<<(ncc + 255) / 256, 256, 0, stream>>>(Wv, wvb, C, C);
    convT_bf16_kernel<<<(ncc + 255) / 256, 256, 0, stream>>>(Wo, wob, C, C);
    convT_bf16_kernel<<<(nch + 255) / 256, 256, 0, stream>>>(W1, w1b, C, Ch);
    convT_bf16_kernel<<<(nch + 255) / 256, 256, 0, stream>>>(W2, w2b, Ch, C);

    // ---- 2. LN1 ----
    ln_rows_kernel<<<(M + 7) / 8, 256, 0, stream>>>(fx, ln1_g, ln1_b, xb, M);

    // ---- 3. QKV GEMMs (WMMA) ----
    dim3 gCC(C / BN, (M + BM - 1) / BM);
    gemm_bf16_wmma<0><<<gCC, 256, 0, stream>>>(xb, wqb, bq, nullptr, nullptr, qb, M, C, C);
    gemm_bf16_wmma<0><<<gCC, 256, 0, stream>>>(xb, wkb, bk, nullptr, nullptr, kb, M, C, C);
    gemm_bf16_wmma<0><<<gCC, 256, 0, stream>>>(xb, wvb, bv, nullptr, nullptr, vb, M, C, C);

    // ---- 4. per-head LN on K, V ----
    int R8 = M * 8;
    head_ln_kernel<<<(R8 + 7) / 8, 256, 0, stream>>>(kb, kln_g, kln_b, R8);
    head_ln_kernel<<<(R8 + 7) / 8, 256, 0, stream>>>(vb, vln_g, vln_b, R8);

    // ---- 5. kv = K^T V (fp32 accum, then 1/N applied in attn) ----
    int nkv = B * 8 * 32 * 32;
    zero_f32_kernel<<<(nkv + 255) / 256, 256, 0, stream>>>(kvf, nkv);
    int S = (Ntok + 127) / 128;
    kv_kernel<<<B * 8 * S, 256, 0, stream>>>(kb, vb, kvf, Ntok, S);

    // ---- 6. attn = q @ kv / N (fused head concat) ----
    size_t tot = (size_t)M * C;
    attn_kernel<<<(int)((tot + 255) / 256), 256, 0, stream>>>(qb, kvf, attnb, Ntok);

    // ---- 7. y1 = fx + attn @ Wo + bo ----
    gemm_bf16_wmma<1><<<gCC, 256, 0, stream>>>(attnb, wob, bo, fx, y1, nullptr, M, C, C);

    // ---- 8. LN2 ----
    ln_rows_kernel<<<(M + 7) / 8, 256, 0, stream>>>(y1, ln2_g, ln2_b, x2b, M);

    // ---- 9. h = gelu(x2 @ W1 + b1) ----
    dim3 gCh(Ch / BN, (M + BM - 1) / BM);
    gemm_bf16_wmma<2><<<gCh, 256, 0, stream>>>(x2b, w1b, b1, nullptr, nullptr, hb, M, C, Ch);

    // ---- 10. out = y1 + h @ W2 + b2 ----
    gemm_bf16_wmma<1><<<gCC, 256, 0, stream>>>(hb, w2b, b2, y1, (float*)d_out, nullptr, M, Ch, C);
}
